// JMEFairnessLoss_28613072126686
// MI455X (gfx1250) — compile-verified, matrix operand order checked by reading, not run on hardware
//
#include <hip/hip_runtime.h>
#include <hip/hip_bf16.h>

#define BATCH        1024
#define NUM_ITEMS    10000
#define TOP_K        100
#define KPAD         128
#define NUM_SAMPLES  10
#define NUM_UG       8
#define NUM_IG       16
#define NUM_PAIRS    (NUM_UG * NUM_IG)
#define LN2          0.6931471806f
#define SIG_SCALE    5.0f                  /* 1/(2*tau): sigmoid = .5+.5*tanh */
#define LOG2_GAMMA   (-0.3219280949f)      /* log2(0.8) */

typedef __attribute__((ext_vector_type(16))) _Float16 v16h;
typedef __attribute__((ext_vector_type(8)))  float    v8f;

__device__ __forceinline__ unsigned hash_u32(unsigned x) {
    x ^= x >> 16; x *= 0x7feb352dU;
    x ^= x >> 15; x *= 0x846ca68bU;
    x ^= x >> 16;
    return x;
}

__device__ __forceinline__ float gumbel_from(unsigned h) {
    // uniform in (0,1) -> standard Gumbel, via v_log_f32 (base-2) + scale
    float u  = ((float)(h >> 8) + 0.5f) * (1.0f / 16777216.0f);
    float v  = -LN2 * __builtin_amdgcn_logf(u);   // -ln(u) > 0
    return -LN2 * __builtin_amdgcn_logf(v);       // -ln(-ln(u))
}

// ---------------------------------------------------------------------------
// Zero the accumulator region of the workspace (harness poisons it to 0xAA).
// ---------------------------------------------------------------------------
__global__ void init_ws_kernel(float* ws) {
    for (int j = threadIdx.x; j < 512; j += blockDim.x) ws[j] = 0.0f;
}

// ---------------------------------------------------------------------------
// Sum of true_scores -> ws[0]. float4 streaming + in-bounds prefetch.
// ---------------------------------------------------------------------------
__global__ void mean_reduce_kernel(const float4* __restrict__ x4,
                                   float* __restrict__ total, int n4) {
    __shared__ float acc;
    if (threadIdx.x == 0) acc = 0.0f;
    __syncthreads();
    const int stride = gridDim.x * blockDim.x;
    float s = 0.0f;
    for (int i = blockIdx.x * blockDim.x + threadIdx.x; i < n4; i += stride) {
        if (i + stride < n4) __builtin_prefetch(&x4[i + stride], 0, 1);
        float4 v = x4[i];
        s += (v.x + v.y) + (v.z + v.w);
    }
    atomicAdd(&acc, s);
    __syncthreads();
    if (threadIdx.x == 0) atomicAdd(total, acc);
}

// ---------------------------------------------------------------------------
// Per-row top-100 via 12-bit monotonic-key histogram select (2 passes).
// ---------------------------------------------------------------------------
__device__ __forceinline__ unsigned mono_key(float f) {
    unsigned u = __float_as_uint(f);
    return u ^ ((u & 0x80000000u) ? 0xFFFFFFFFu : 0x80000000u);
}

__global__ void topk_kernel(const float* __restrict__ pred,
                            float* __restrict__ tk_val,
                            int* __restrict__ tk_idx) {
    __shared__ unsigned hist[4096];
    __shared__ unsigned chunk_sum[256];
    __shared__ int sh_T, sh_above;
    __shared__ unsigned cnt_hi, cnt_eq;

    const int row = blockIdx.x;
    const float4* p4 = (const float4*)(pred + (size_t)row * NUM_ITEMS);
    const int n4 = NUM_ITEMS / 4;   // 2500

    for (int i = threadIdx.x; i < 4096; i += blockDim.x) hist[i] = 0u;
    if (threadIdx.x == 0) { cnt_hi = 0u; cnt_eq = 0u; }
    __syncthreads();

    for (int i = threadIdx.x; i < n4; i += blockDim.x) {
        float4 v = p4[i];
        atomicAdd(&hist[mono_key(v.x) >> 20], 1u);
        atomicAdd(&hist[mono_key(v.y) >> 20], 1u);
        atomicAdd(&hist[mono_key(v.z) >> 20], 1u);
        atomicAdd(&hist[mono_key(v.w) >> 20], 1u);
    }
    __syncthreads();

    // two-level descending scan: 256 chunks of 16 buckets
    {
        unsigned s = 0;
        int base = threadIdx.x * 16;
#pragma unroll
        for (int j = 0; j < 16; ++j) s += hist[base + j];
        chunk_sum[threadIdx.x] = s;
    }
    __syncthreads();
    if (threadIdx.x == 0) {
        int cum = 0, c = 255;
        for (; c > 0; --c) {
            int cs = (int)chunk_sum[c];
            if (cum + cs >= TOP_K) break;
            cum += cs;
        }
        int t = c * 16 + 15;
        for (; t > c * 16; --t) {
            int h = (int)hist[t];
            if (cum + h >= TOP_K) break;
            cum += h;
        }
        sh_T = t;
        sh_above = cum;   // exact count of elements strictly above bucket T
    }
    __syncthreads();

    const int T = sh_T, above = sh_above;
    for (int i = threadIdx.x; i < n4; i += blockDim.x) {
        float4 v = p4[i];
        float vv[4] = {v.x, v.y, v.z, v.w};
#pragma unroll
        for (int c = 0; c < 4; ++c) {
            float f = vv[c];
            int b = (int)(mono_key(f) >> 20);
            int slot = -1;
            if (b > T) {
                slot = (int)atomicAdd(&cnt_hi, 1u);       // [0, above)
            } else if (b == T) {
                int e = (int)atomicAdd(&cnt_eq, 1u);
                int s2 = above + e;
                if (s2 < TOP_K) slot = s2;                // fill remainder
            }
            if (slot >= 0 && slot < TOP_K) {
                tk_val[row * TOP_K + slot] = f;
                tk_idx[row * TOP_K + slot] = 4 * i + c;
            }
        }
    }
}

// ---------------------------------------------------------------------------
// Exposure + both losses. One block per batch row; 8 waves; wave w owns rows
// [16w,16w+16) of the 128x128 pairwise matrix.
//
// sigmoid((ni-nj)/tau) = 0.5 + 0.5*tanh((ni-nj)/(2*tau)); we feed raw
// tanh values into WMMA (A = tanh tile, B = ones), so
//   sum_j sigmoid = 64 + 0.5 * T      (T = WMMA row sum over 128 padded cols;
//                                      padding -> tanh = -1 -> sigmoid 0)
//   rank - 1      = 63.5 + 0.5 * T
// Per matrix element: v_sub + v_tanh_f32 + 0.5 * v_cvt_pk_f16_f32.
// ---------------------------------------------------------------------------
__global__ void exposure_loss_kernel(const float* __restrict__ tk_val,
                                     const int* __restrict__ tk_idx,
                                     const int* __restrict__ user_groups,
                                     const int* __restrict__ item_groups,
                                     const float* __restrict__ true_sum,
                                     float* __restrict__ ii_sum,
                                     float* __restrict__ pair_sums,
                                     float* __restrict__ pair_cnts) {
    __shared__ __align__(16) float noisy[KPAD];   // pre-scaled by SIG_SCALE
    __shared__ float expo[KPAD];
    __shared__ float gsum[NUM_IG];
    __shared__ float gcnt[NUM_IG];
    __shared__ float ii_acc;

    const int row  = blockIdx.x;
    const int tid  = threadIdx.x;
    const int lane = tid & 31;
    const int wv   = tid >> 5;            // wave 0..7 owns rows [16wv,16wv+16)
    const int hi   = (lane >= 16) ? 8 : 0;

    if (tid < KPAD) expo[tid] = 0.0f;
    if (tid < NUM_IG) { gsum[tid] = 0.0f; gcnt[tid] = 0.0f; }
    if (tid == 0) ii_acc = 0.0f;

    v16h bones;
#pragma unroll
    for (int e = 0; e < 16; ++e) bones[e] = (_Float16)1.0f;

    for (int s = 0; s < NUM_SAMPLES; ++s) {
        __syncthreads();                  // orders expo/noisy across iterations
        if (tid < KPAD) {
            float v;
            if (tid < TOP_K) {
                unsigned h = hash_u32(
                    ((unsigned)((s * BATCH + row) * KPAD + tid)) ^ 0x9E3779B9u);
                v = (tk_val[row * TOP_K + tid] + gumbel_from(h)) * SIG_SCALE;
            } else {
                v = 1.0e31f;              // padded cols: tanh -> -1, sig -> 0
            }
            noisy[tid] = v;
        }
        __syncthreads();

        const int   irow = (wv << 4) + (lane & 15);
        const float ni   = noisy[irow];
        v8f acc = {};
#pragma unroll
        for (int chunk = 0; chunk < 4; ++chunk) {
            // 16-bit A 16x32 fragment (ISA 7.12.2): this lane needs K-runs
            // [b0, b0+7] (elements 0..7) and [b0+16, b0+23] (elements 8..15).
            const int b0 = hi + (chunk << 5);
            float4 r0 = *(const float4*)&noisy[b0];
            float4 r1 = *(const float4*)&noisy[b0 + 4];
            float4 r2 = *(const float4*)&noisy[b0 + 16];
            float4 r3 = *(const float4*)&noisy[b0 + 20];
            float nj[16] = {r0.x, r0.y, r0.z, r0.w, r1.x, r1.y, r1.z, r1.w,
                            r2.x, r2.y, r2.z, r2.w, r3.x, r3.y, r3.z, r3.w};
            v16h a;
#pragma unroll
            for (int e = 0; e < 16; ++e) {
                float t = ni - nj[e];                      // (ni-nj)/(2*tau)
                a[e] = (_Float16)__builtin_amdgcn_tanhf(t);
            }
            acc = __builtin_amdgcn_wmma_f32_16x16x32_f16(
                false, a, false, bones, (short)0, acc, false, false);
        }
        // lanes with N==0 (lane 0 / lane 16) hold rows base+d in acc[d]
        if ((lane & 15) == 0) {
            int base = (wv << 4) + hi;
#pragma unroll
            for (int d = 0; d < 8; ++d) {
                float rank_m1 = fmaf(0.5f, acc[d], 63.5f);
                expo[base + d] += __builtin_amdgcn_exp2f(LOG2_GAMMA * rank_m1);
            }
        }
    }
    __syncthreads();

    const float target =
        true_sum[0] * (1.0f / ((float)BATCH * (float)NUM_ITEMS));
    const float inv_s = 1.0f / (float)NUM_SAMPLES;

    float d2 = 0.0f;
    if (tid < TOP_K) {
        float delta = expo[tid] * inv_s - target;
        d2 = delta * delta;
        int ig = item_groups[tk_idx[row * TOP_K + tid]] & (NUM_IG - 1);
        atomicAdd(&gsum[ig], delta);
        atomicAdd(&gcnt[ig], 1.0f);
    }
    atomicAdd(&ii_acc, d2);
    __syncthreads();

    if (tid == 0) atomicAdd(ii_sum, ii_acc);
    if (tid < NUM_IG) {
        int ug = user_groups[row] & (NUM_UG - 1);
        if (gcnt[tid] > 0.0f) {
            atomicAdd(&pair_sums[ug * NUM_IG + tid], gsum[tid]);
            atomicAdd(&pair_cnts[ug * NUM_IG + tid], gcnt[tid]);
        }
    }
}

// ---------------------------------------------------------------------------
// Final scalar combine: (total, ii, gg)
// ---------------------------------------------------------------------------
__global__ void finalize_kernel(const float* __restrict__ ii_sum,
                                const float* __restrict__ pair_sums,
                                const float* __restrict__ pair_cnts,
                                float* __restrict__ out) {
    __shared__ float acc;
    if (threadIdx.x == 0) acc = 0.0f;
    __syncthreads();
    if (threadIdx.x < NUM_PAIRS) {
        float c = pair_cnts[threadIdx.x];
        float s = pair_sums[threadIdx.x];
        float avg = (c > 0.0f) ? (s / fmaxf(c, 1.0f)) : 0.0f;
        atomicAdd(&acc, avg * avg);
    }
    __syncthreads();
    if (threadIdx.x == 0) {
        float gg = acc / (float)NUM_PAIRS;
        float ii = ii_sum[0] / ((float)BATCH * (float)TOP_K);
        out[0] = ii + 1.0f * gg;   // ALPHA = 1.0
        out[1] = ii;
        out[2] = gg;
    }
}

// ---------------------------------------------------------------------------
extern "C" void kernel_launch(void* const* d_in, const int* in_sizes, int n_in,
                              void* d_out, int out_size, void* d_ws,
                              size_t ws_size, hipStream_t stream) {
    const float* pred  = (const float*)d_in[0];
    const float* truev = (const float*)d_in[1];
    const int*   ug    = (const int*)d_in[2];
    const int*   ig    = (const int*)d_in[3];
    float* out = (float*)d_out;

    // Workspace layout (floats):
    //  [0]        true_sum
    //  [1]        ii_sum
    //  [16..143]  pair_sums[128]
    //  [144..271] pair_cnts[128]
    //  [512..]    topk values (1024*100 f32), then topk indices (1024*100 i32)
    float* wsf       = (float*)d_ws;
    float* true_sum  = wsf + 0;
    float* ii_sum    = wsf + 1;
    float* pair_sums = wsf + 16;
    float* pair_cnts = wsf + 144;
    float* tk_val    = wsf + 512;
    int*   tk_idx    = (int*)(wsf + 512 + BATCH * TOP_K);

    hipLaunchKernelGGL(init_ws_kernel, dim3(1), dim3(256), 0, stream, wsf);
    hipLaunchKernelGGL(mean_reduce_kernel, dim3(2048), dim3(256), 0, stream,
                       (const float4*)truev, true_sum,
                       (BATCH * NUM_ITEMS) / 4);
    hipLaunchKernelGGL(topk_kernel, dim3(BATCH), dim3(256), 0, stream,
                       pred, tk_val, tk_idx);
    hipLaunchKernelGGL(exposure_loss_kernel, dim3(BATCH), dim3(256), 0, stream,
                       tk_val, tk_idx, ug, ig, true_sum, ii_sum, pair_sums,
                       pair_cnts);
    hipLaunchKernelGGL(finalize_kernel, dim3(1), dim3(128), 0, stream,
                       ii_sum, pair_sums, pair_cnts, out);
}
